// NER_76012331205088
// MI455X (gfx1250) — compile-verified, matrix operand order checked by reading, not run on hardware
//
#include <hip/hip_runtime.h>
#include <hip/hip_bf16.h>
#include <stdint.h>

// ---------------- problem constants ----------------
#define Bc    16
#define Sc    2048
#define Hc    768
#define H3c   2304
#define Ec    128
#define TAGc  10
// end_logit scale: mean over 12 heads then /sqrt(64) -> /96
#define INV_SCALE (1.0f / 96.0f)

// LDS stage geometry: 64 K-halves per stage, rows padded to 72 halves (144B =
// 36 dwords; 36r mod 64 hits 16 distinct 4-bank groups -> conflict-free b128)
#define KS_HALVES   64
#define ROW_STRIDE  72
#define NSTAGES     (Hc / KS_HALVES)   // 12

typedef __attribute__((ext_vector_type(16))) __bf16 v16bf;
typedef __attribute__((ext_vector_type(8)))  float  v8f;

union Frag {
    uint4 q[2];
    v16bf v;
};

static __device__ __forceinline__ v8f wmma_bf16(const Frag& a, const Frag& b, v8f c) {
    return __builtin_amdgcn_wmma_f32_16x16x32_bf16(
        /*neg_a=*/false, a.v, /*neg_b=*/false, b.v,
        /*c_mod=*/(short)0, c, /*reuse_a=*/false, /*reuse_b=*/false);
}

static __device__ __forceinline__ unsigned short f2bf(float f) {
    unsigned u = __builtin_bit_cast(unsigned, f);
    unsigned r = u + 0x7FFFu + ((u >> 16) & 1u);   // RNE
    return (unsigned short)(r >> 16);
}

// CDNA5 async copy: global -> LDS, 16B per lane, tracked by ASYNCcnt.
static __device__ __forceinline__ void async_cp16(unsigned lds_byte,
                                                  const unsigned short* g) {
    asm volatile("global_load_async_to_lds_b128 %0, %1, off"
                 :: "v"(lds_byte), "v"(g) : "memory");
}
static __device__ __forceinline__ void wait_async0() {
    asm volatile("s_wait_asynccnt 0x0" ::: "memory");
}
static __device__ __forceinline__ unsigned lds_off(const void* p) {
    // generic LDS pointer = {shared aperture, 32-bit LDS offset}
    return (unsigned)(uintptr_t)p;
}

// ---------------- fp32 -> bf16 convert (4 elems/thread) ----------------
__global__ __launch_bounds__(256) void cvt_bf16(const float* __restrict__ in,
                                                unsigned short* __restrict__ out,
                                                int n4) {
    int i = blockIdx.x * blockDim.x + threadIdx.x;
    if (i >= n4) return;
    int idx = i * 4;
    float4 f = *(const float4*)(in + idx);
    unsigned long long p = (unsigned long long)f2bf(f.x)
                         | ((unsigned long long)f2bf(f.y) << 16)
                         | ((unsigned long long)f2bf(f.z) << 32)
                         | ((unsigned long long)f2bf(f.w) << 48);
    *(unsigned long long*)(out + idx) = p;
}

// o_w [TAG,3H] -> owt [3H][16] (padded rows, vector-loadable in the epilogue)
__global__ __launch_bounds__(256) void transpose_ow(const float* __restrict__ ow,
                                                    float* __restrict__ owt) {
    int n = blockIdx.x * blockDim.x + threadIdx.x;
    if (n >= H3c) return;
#pragma unroll
    for (int t = 0; t < TAGc; ++t) owt[(size_t)n * 16 + t] = ow[(size_t)t * H3c + n];
#pragma unroll
    for (int t = TAGc; t < 16; ++t) owt[(size_t)n * 16 + t] = 0.0f;
}

// A fragment straight from row-major global bf16 (base already + half*8 halves)
static __device__ __forceinline__ Frag load_fragA(const unsigned short* base_half,
                                                  int k0_half) {
    Frag f;
    const uint4* p = (const uint4*)(base_half + k0_half);
    f.q[0] = p[0];
    f.q[1] = p[2];
    return f;
}
// B fragment from an LDS-staged row (row already selected; k0 = 0 or 32)
static __device__ __forceinline__ Frag load_fragB_lds(const unsigned short* row_base,
                                                      int k0, int half) {
    Frag f;
    f.q[0] = *(const uint4*)(row_base + k0 + half * 8);
    f.q[1] = *(const uint4*)(row_base + k0 + half * 8 + 16);
    return f;
}

// ---------------- fused GLU start-logit kernel ----------------
// 256 thr = 8 waves; wave owns a 16-row M tile (block = 128 rows of B*S).
// N chunk = 32 cols of u AND v. Stage = 32 rows x 64 K-halves of u+v (8KB),
// async double-buffered in LDS, shared by all 8 waves; 8 WMMAs per wave per
// stage, one barrier per stage. Epilogue: glu = sigmoid(u)*v projected onto
// transposed o_w into TAG regs, reduced across lanes with ds_add_f32.
__global__ __launch_bounds__(256) void glu_start_kernel(
    const unsigned short* __restrict__ hs,   // [B*S, H] bf16
    const unsigned short* __restrict__ uw,   // [3H, H] bf16
    const unsigned short* __restrict__ vw,   // [3H, H] bf16
    const float* __restrict__ ub, const float* __restrict__ vb,
    const float* __restrict__ owt,           // [3H, 16] f32 (transposed o_w)
    const float* __restrict__ ob,
    float* __restrict__ out)                 // [B*S, TAG]
{
    __shared__ __align__(16) unsigned short wtile[2][2][32][ROW_STRIDE];
    __shared__ float tacc[128 * TAGc];

    int tid = threadIdx.x;
    for (int i = tid; i < 128 * TAGc; i += 256) tacc[i] = 0.0f;

    int wave = tid >> 5, lane = tid & 31;
    int half = lane >> 4, r = lane & 15;
    int rowBase = blockIdx.x * 128 + wave * 16;
    const unsigned short* Abase = hs + (size_t)(rowBase + r) * Hc + half * 8;

    // cooperative-load mapping: 512 x 16B segments/stage; thread does 1 u + 1 v
    int lrow = (tid >> 3) & 31;   // 0..31
    int lseg = tid & 7;           // 16B segment within 64-half row

    float tagacc[8][TAGc];
#pragma unroll
    for (int rr = 0; rr < 8; ++rr)
#pragma unroll
        for (int t = 0; t < TAGc; ++t) tagacc[rr][t] = 0.0f;

    __syncthreads();

    for (int nt = 0; nt < H3c / 16; nt += 2) {
        const unsigned short* usrc = uw + (size_t)(nt * 16 + lrow) * Hc + lseg * 8;
        const unsigned short* vsrc = vw + (size_t)(nt * 16 + lrow) * Hc + lseg * 8;
        unsigned ldsu0 = lds_off(&wtile[0][0][lrow][lseg * 8]);
        unsigned ldsv0 = lds_off(&wtile[0][1][lrow][lseg * 8]);
        unsigned ldsu1 = lds_off(&wtile[1][0][lrow][lseg * 8]);
        unsigned ldsv1 = lds_off(&wtile[1][1][lrow][lseg * 8]);

        // prologue: stage 0
        async_cp16(ldsu0, usrc);
        async_cp16(ldsv0, vsrc);
        wait_async0();
        __syncthreads();

        v8f accu[2] = { {}, {} };
        v8f accv[2] = { {}, {} };

        for (int ks = 0; ks < NSTAGES; ++ks) {
            int cur = ks & 1;
            if (ks + 1 < NSTAGES) {   // prefetch next stage while computing
                async_cp16(cur ? ldsu0 : ldsu1, usrc + (ks + 1) * KS_HALVES);
                async_cp16(cur ? ldsv0 : ldsv1, vsrc + (ks + 1) * KS_HALVES);
            }
#pragma unroll
            for (int ktl = 0; ktl < 2; ++ktl) {
                Frag A = load_fragA(Abase, ks * KS_HALVES + ktl * 32);
#pragma unroll
                for (int j = 0; j < 2; ++j) {
                    Frag Bu = load_fragB_lds(&wtile[cur][0][j * 16 + r][0], ktl * 32, half);
                    accu[j] = wmma_bf16(A, Bu, accu[j]);
                    Frag Bv = load_fragB_lds(&wtile[cur][1][j * 16 + r][0], ktl * 32, half);
                    accv[j] = wmma_bf16(A, Bv, accv[j]);
                }
            }
            wait_async0();
            __syncthreads();
        }

        // epilogue: GLU + project onto transposed o_w for this lane's column n
#pragma unroll
        for (int j = 0; j < 2; ++j) {
            int n = (nt + j) * 16 + r;
            float ubias = ub[n];
            float vbias = vb[n];
            const float* owr = owt + (size_t)n * 16;
            float4 w0 = *(const float4*)(owr);
            float4 w1 = *(const float4*)(owr + 4);
            float2 w2 = *(const float2*)(owr + 8);
            float owv[TAGc] = { w0.x, w0.y, w0.z, w0.w,
                                w1.x, w1.y, w1.z, w1.w,
                                w2.x, w2.y };
#pragma unroll
            for (int rr = 0; rr < 8; ++rr) {
                float uu = accu[j][rr] + ubias;
                float vv = accv[j][rr] + vbias;
                float g  = vv / (1.0f + __expf(-uu));
#pragma unroll
                for (int t = 0; t < TAGc; ++t) tagacc[rr][t] += g * owv[t];
            }
        }
    }

    // reduce partial TAG sums across the 16 lanes of each half via LDS atomics
#pragma unroll
    for (int rr = 0; rr < 8; ++rr) {
        int m = wave * 16 + rr + half * 8;
#pragma unroll
        for (int t = 0; t < TAGc; ++t)
            atomicAdd(&tacc[m * TAGc + t], tagacc[rr][t]);
    }
    __syncthreads();

    for (int i = tid; i < 128 * TAGc; i += 256) {
        int m = i / TAGc, t = i % TAGc;
        out[(size_t)(blockIdx.x * 128 + m) * TAGc + t] = tacc[i] + ob[t];
    }
}

// ---------------- K projection: Kmat = bf16(hs @ k_w.T + k_b) ----------------
// Same async-LDS staging: weight tile shared by all 8 waves.
__global__ __launch_bounds__(256) void proj_k_kernel(
    const unsigned short* __restrict__ hs,
    const unsigned short* __restrict__ kw,
    const float* __restrict__ kb,
    unsigned short* __restrict__ kout)
{
    __shared__ __align__(16) unsigned short wtile[2][32][ROW_STRIDE];

    int tid = threadIdx.x;
    int wave = tid >> 5, lane = tid & 31;
    int half = lane >> 4, r = lane & 15;
    int rowBase = blockIdx.x * 128 + wave * 16;
    const unsigned short* Abase = hs + (size_t)(rowBase + r) * Hc + half * 8;

    // 256 x 16B segments per stage; one per thread
    int lrow = tid >> 3;     // 0..31
    int lseg = tid & 7;

    for (int nt = 0; nt < Hc / 16; nt += 2) {
        const unsigned short* ksrc = kw + (size_t)(nt * 16 + lrow) * Hc + lseg * 8;
        unsigned lds0 = lds_off(&wtile[0][lrow][lseg * 8]);
        unsigned lds1 = lds_off(&wtile[1][lrow][lseg * 8]);

        async_cp16(lds0, ksrc);
        wait_async0();
        __syncthreads();

        v8f acc[2] = { {}, {} };
        for (int ks = 0; ks < NSTAGES; ++ks) {
            int cur = ks & 1;
            if (ks + 1 < NSTAGES)
                async_cp16(cur ? lds0 : lds1, ksrc + (ks + 1) * KS_HALVES);
#pragma unroll
            for (int ktl = 0; ktl < 2; ++ktl) {
                Frag A = load_fragA(Abase, ks * KS_HALVES + ktl * 32);
#pragma unroll
                for (int j = 0; j < 2; ++j) {
                    Frag Bf = load_fragB_lds(&wtile[cur][j * 16 + r][0], ktl * 32, half);
                    acc[j] = wmma_bf16(A, Bf, acc[j]);
                }
            }
            wait_async0();
            __syncthreads();
        }
#pragma unroll
        for (int j = 0; j < 2; ++j) {
            int n = (nt + j) * 16 + r;
            float bias = kb[n];
#pragma unroll
            for (int rr = 0; rr < 8; ++rr) {
                int row = rowBase + rr + half * 8;
                kout[(size_t)row * Hc + n] = f2bf(acc[j][rr] + bias);
            }
        }
    }
}

// ---------------- Q projection with entity gather (small) ----------------
__global__ __launch_bounds__(256) void proj_q_kernel(
    const unsigned short* __restrict__ hs,
    const long long* __restrict__ ent,
    const unsigned short* __restrict__ qw,
    const float* __restrict__ qb,
    unsigned short* __restrict__ qout)
{
    int b = blockIdx.x;
    int tid = threadIdx.x;
    int wave = tid >> 5, lane = tid & 31;
    int half = lane >> 4, r = lane & 15;

    int e = wave * 16 + r;
    long long srow = ent[(size_t)b * Ec + e];
    const unsigned short* Abase = hs + ((size_t)b * Sc + (size_t)srow) * Hc + half * 8;
    int outRowBase = b * Ec + wave * 16;

    for (int nt = 0; nt < Hc / 16; nt += 2) {
        v8f acc[2] = { {}, {} };
        const unsigned short* B0 = qw + (size_t)(nt * 16 + r) * Hc + half * 8;
        for (int kt = 0; kt < Hc / 32; ++kt) {
            Frag A = load_fragA(Abase, kt * 32);
#pragma unroll
            for (int j = 0; j < 2; ++j) {
                Frag Bf = load_fragA(B0 + (size_t)j * 16 * Hc, kt * 32);
                acc[j] = wmma_bf16(A, Bf, acc[j]);
            }
        }
#pragma unroll
        for (int j = 0; j < 2; ++j) {
            int n = (nt + j) * 16 + r;
            float bias = qb[n];
#pragma unroll
            for (int rr = 0; rr < 8; ++rr) {
                int row = outRowBase + rr + half * 8;
                qout[(size_t)row * Hc + n] = f2bf(acc[j][rr] + bias);
            }
        }
    }
}

// ---------------- scores = Q @ K^T / 96, masked ----------------
// grid.x = B * (S/128). All 8 waves share the 128-column K^T stage tile
// (128 rows x 64 K-halves, 16KB, double buffered); 16 WMMAs per wave/stage.
__global__ __launch_bounds__(256) void scores_kernel(
    const unsigned short* __restrict__ qm,    // [B*E, H] bf16
    const unsigned short* __restrict__ km,    // [B*S, H] bf16
    const unsigned char* __restrict__ mask,   // [B, S] bool
    float* __restrict__ out)                  // [B, E, S]
{
    __shared__ __align__(16) unsigned short ktile[2][128][ROW_STRIDE];

    int b = blockIdx.x >> 4;
    int nchunk = blockIdx.x & 15;
    int tid = threadIdx.x;
    int wave = tid >> 5, lane = tid & 31;
    int half = lane >> 4, r = lane & 15;

    const unsigned short* Abase = qm + ((size_t)b * Ec + wave * 16 + r) * Hc + half * 8;
    int nbase0 = nchunk * 128;

    // 1024 x 16B segments per stage; 4 per thread
    const unsigned short* ksrc[4];
    unsigned lstage0[4], lstage1[4];
#pragma unroll
    for (int i = 0; i < 4; ++i) {
        int sid = tid + i * 256;
        int row = sid >> 3, seg = sid & 7;
        ksrc[i] = km + ((size_t)b * Sc + nbase0 + row) * Hc + seg * 8;
        lstage0[i] = lds_off(&ktile[0][row][seg * 8]);
        lstage1[i] = lds_off(&ktile[1][row][seg * 8]);
    }

    // prologue: stage 0
#pragma unroll
    for (int i = 0; i < 4; ++i) async_cp16(lstage0[i], ksrc[i]);
    wait_async0();
    __syncthreads();

    v8f acc[8] = { {}, {}, {}, {}, {}, {}, {}, {} };

    for (int ks = 0; ks < NSTAGES; ++ks) {
        int cur = ks & 1;
        if (ks + 1 < NSTAGES) {
#pragma unroll
            for (int i = 0; i < 4; ++i)
                async_cp16(cur ? lstage0[i] : lstage1[i], ksrc[i] + (ks + 1) * KS_HALVES);
        }
#pragma unroll
        for (int ktl = 0; ktl < 2; ++ktl) {
            Frag A = load_fragA(Abase, ks * KS_HALVES + ktl * 32);
#pragma unroll
            for (int j = 0; j < 8; ++j) {
                Frag Bf = load_fragB_lds(&ktile[cur][j * 16 + r][0], ktl * 32, half);
                acc[j] = wmma_bf16(A, Bf, acc[j]);
            }
        }
        wait_async0();
        __syncthreads();
    }

#pragma unroll
    for (int j = 0; j < 8; ++j) {
        int s = nbase0 + j * 16 + r;
        bool ok = mask[(size_t)b * Sc + s] != 0;
#pragma unroll
        for (int rr = 0; rr < 8; ++rr) {
            int e = wave * 16 + rr + half * 8;
            float val = ok ? acc[j][rr] * INV_SCALE : -50000.0f;
            out[((size_t)b * Ec + e) * Sc + s] = val;
        }
    }
}

// ---------------- launcher ----------------
extern "C" void kernel_launch(void* const* d_in, const int* in_sizes, int n_in,
                              void* d_out, int out_size, void* d_ws, size_t ws_size,
                              hipStream_t stream) {
    const float*     hs_f   = (const float*)d_in[0];
    const float*     u_w    = (const float*)d_in[1];
    const float*     u_b    = (const float*)d_in[2];
    const float*     v_w    = (const float*)d_in[3];
    const float*     v_b    = (const float*)d_in[4];
    const float*     o_w    = (const float*)d_in[5];
    const float*     o_b    = (const float*)d_in[6];
    const float*     q_w    = (const float*)d_in[7];
    const float*     q_b    = (const float*)d_in[8];
    const float*     k_w    = (const float*)d_in[9];
    const float*     k_b    = (const float*)d_in[10];
    const long long* ent    = (const long long*)d_in[11];
    const unsigned char* am = (const unsigned char*)d_in[12];

    float* out_start = (float*)d_out;                           // B*S*TAG
    float* out_end   = out_start + (size_t)Bc * Sc * TAGc;      // B*E*S

    char* ws = (char*)d_ws;
    unsigned short* hs_bf = (unsigned short*)(ws);
    size_t off = (size_t)Bc * Sc * Hc * 2;
    unsigned short* uw_bf = (unsigned short*)(ws + off);   off += (size_t)H3c * Hc * 2;
    unsigned short* vw_bf = (unsigned short*)(ws + off);   off += (size_t)H3c * Hc * 2;
    unsigned short* kw_bf = (unsigned short*)(ws + off);   off += (size_t)Hc * Hc * 2;
    unsigned short* qw_bf = (unsigned short*)(ws + off);   off += (size_t)Hc * Hc * 2;
    unsigned short* km_bf = (unsigned short*)(ws + off);   off += (size_t)Bc * Sc * Hc * 2;
    unsigned short* qm_bf = (unsigned short*)(ws + off);   off += (size_t)Bc * Ec * Hc * 2;
    float*          owt   = (float*)(ws + off);            off += (size_t)H3c * 16 * 4;
    (void)ws_size; (void)n_in; (void)in_sizes; (void)out_size;

    {
        int n4 = (Bc * Sc * Hc) / 4;
        cvt_bf16<<<(n4 + 255) / 256, 256, 0, stream>>>(hs_f, hs_bf, n4);
        n4 = (H3c * Hc) / 4;
        cvt_bf16<<<(n4 + 255) / 256, 256, 0, stream>>>(u_w, uw_bf, n4);
        cvt_bf16<<<(n4 + 255) / 256, 256, 0, stream>>>(v_w, vw_bf, n4);
        n4 = (Hc * Hc) / 4;
        cvt_bf16<<<(n4 + 255) / 256, 256, 0, stream>>>(k_w, kw_bf, n4);
        cvt_bf16<<<(n4 + 255) / 256, 256, 0, stream>>>(q_w, qw_bf, n4);
        transpose_ow<<<(H3c + 255) / 256, 256, 0, stream>>>(o_w, owt);
    }

    glu_start_kernel<<<(Bc * Sc) / 128, 256, 0, stream>>>(
        hs_bf, uw_bf, vw_bf, u_b, v_b, owt, o_b, out_start);

    proj_k_kernel<<<(Bc * Sc) / 128, 256, 0, stream>>>(hs_bf, kw_bf, k_b, km_bf);

    proj_q_kernel<<<Bc, 256, 0, stream>>>(hs_bf, ent, qw_bf, q_b, qm_bf);

    scores_kernel<<<Bc * (Sc / 128), 256, 0, stream>>>(qm_bf, km_bf, am, out_end);
}
